// HGT_54649163874901
// MI455X (gfx1250) — matrix operand order repaired and checked by reading, not compiled, not run on hardware
//
#include <hip/hip_runtime.h>
#include <hip/hip_bf16.h>

// ---------------------------------------------------------------------------
// HGT forward for MI455X (gfx1250, wave32, WMMA + pipelined async-to-LDS).
//  - All dense math: v_wmma_f32_16x16x32_f16, f32 accumulate.
//  - A operands f16 in global (halves dominant read traffic); B operands
//    pre-transposed f16 [N][K]. Both staged with global_load_async_to_lds_b128
//    into DOUBLE-BUFFERED LDS; in-order ASYNCcnt retirement lets us overlap
//    tile k+1 staging with tile k compute using threshold waits.
//  - Edge attention: bandwidth-bound VALU gather/scatter with f32 atomics.
// ---------------------------------------------------------------------------

typedef _Float16 v16h __attribute__((ext_vector_type(16)));
typedef float    v8f  __attribute__((ext_vector_type(8)));
typedef _Float16 h4   __attribute__((ext_vector_type(4)));

union HVec { v16h v; uint4 q[2]; _Float16 h[16]; };

#define BM 256
#define BN 64
#define BK 32
#define KPAD 8   // row pitch 40 halves = 80B (multiple of 16B)

__device__ __forceinline__ float gelu_exact(float x) {
    return 0.5f * x * (1.0f + erff(x * 0.70710678118654752f));
}

// ---------------------------------------------------------------------------
// f16 WMMA GEMM, software-pipelined async staging.
//   C[M,N] = epilogue(A[M,K](f16) @ Bt[N,K]^T(f16) + bias)
//   reluC: ReLU.  Xold!=null: C = sk*C + (1-sk)*Xold, sk=sigmoid(*skipGate).
//   Cf (f32) and/or Ch (f16) destinations.
// Requires K % 32 == 0, N % 64 == 0; M guarded.
// 256 threads = 8 waves; wave w owns rows [w*32, w*32+32) x 64 cols
// -> 8 WMMA per K-step, each B fragment reused twice.
// ---------------------------------------------------------------------------
__global__ __launch_bounds__(256)
void gemm_wmma(const _Float16* __restrict__ A, const _Float16* __restrict__ Bt,
               const float* __restrict__ bias,
               float* __restrict__ Cf, _Float16* __restrict__ Ch,
               const float* __restrict__ Xold, const float* __restrict__ skipGate,
               int M, int N, int K, int reluC)
{
    __shared__ _Float16 sA[2][BM][BK + KPAD];   // [buf][row][k]   2 x 20 KB
    __shared__ _Float16 sB[2][BN][BK + KPAD];   // [buf][col][k]   2 x  5 KB

    const int tid  = threadIdx.x;
    const int lane = tid & 31;
    const int wave = tid >> 5;
    const int hf   = (lane >> 4) & 1;
    const int r16  = lane & 15;

    const int m0 = blockIdx.x * BM;
    const int n0 = blockIdx.y * BN;

    v8f acc[8] = {};

    // A staging: thread t streams row t (64B) of the 256x32 tile.
    const bool aInb = (m0 + tid) < M;
    const bool anyA = __any((int)aInb) != 0;       // uniform per wave
    const _Float16* aRow = A + (size_t)(m0 + tid) * K;
    const unsigned ldsA0 = (unsigned)(size_t)&sA[0][tid][0];
    const unsigned ldsA1 = (unsigned)(size_t)&sA[1][tid][0];

    // B staging: thread t streams 16B of col t>>2.
    const int bcol = tid >> 2;
    const int bseg = (tid & 3) * 8;                 // halves
    const _Float16* bRow = Bt + (size_t)(n0 + bcol) * K + bseg;
    const unsigned ldsB0 = (unsigned)(size_t)&sB[0][bcol][bseg];
    const unsigned ldsB1 = (unsigned)(size_t)&sB[1][bcol][bseg];

    auto issue_tile = [&](int k0, unsigned la, unsigned lb) {
        if (aInb) {
            const _Float16* ga = aRow + k0;
            asm volatile(
                "global_load_async_to_lds_b128 %0, %1, off\n\t"
                "global_load_async_to_lds_b128 %0, %1, off offset:16\n\t"
                "global_load_async_to_lds_b128 %0, %1, off offset:32\n\t"
                "global_load_async_to_lds_b128 %0, %1, off offset:48"
                :: "v"(la), "v"(ga) : "memory");
        }
        const _Float16* gb = bRow + k0;
        asm volatile(
            "global_load_async_to_lds_b128 %0, %1, off"
            :: "v"(lb), "v"(gb) : "memory");
    };

    issue_tile(0, ldsA0, ldsB0);                   // prologue: tile 0 -> buf 0

    int p = 0;
    for (int k0 = 0; k0 < K; k0 += BK, p ^= 1) {
        const bool hasNext = (k0 + BK) < K;
        if (hasNext)                                // tile k+1 -> other buffer
            issue_tile(k0 + BK, p ? ldsA0 : ldsA1, p ? ldsB0 : ldsB1);

        // In-order retirement: cnt <= (#ops just issued) => tile k landed.
        if (hasNext) {
            if (anyA) asm volatile("s_wait_asynccnt 0x5" ::: "memory");
            else      asm volatile("s_wait_asynccnt 0x1" ::: "memory");
        } else {
            asm volatile("s_wait_asynccnt 0x0" ::: "memory");
        }
        __syncthreads();                            // all waves staged buf p

        const _Float16 (*cA)[BK + KPAD] = sA[p];
        const _Float16 (*cB)[BK + KPAD] = sB[p];
        const int ar = wave * 32 + r16;
        HVec ua0, ua1;
        ua0.q[0] = *(const uint4*)&cA[ar][hf * 8];          // K = hf*8 + 0..7
        ua0.q[1] = *(const uint4*)&cA[ar][16 + hf * 8];     // K = 16 + hf*8 + 0..7
        ua1.q[0] = *(const uint4*)&cA[ar + 16][hf * 8];
        ua1.q[1] = *(const uint4*)&cA[ar + 16][16 + hf * 8];
        #pragma unroll
        for (int j = 0; j < 4; ++j) {
            HVec ub;
            ub.q[0] = *(const uint4*)&cB[j * 16 + r16][hf * 8];
            ub.q[1] = *(const uint4*)&cB[j * 16 + r16][16 + hf * 8];
            acc[j]     = __builtin_amdgcn_wmma_f32_16x16x32_f16(
                             false, ua0.v, false, ub.v, (short)0, acc[j],     false, false);
            acc[4 + j] = __builtin_amdgcn_wmma_f32_16x16x32_f16(
                             false, ua1.v, false, ub.v, (short)0, acc[4 + j], false, false);
        }
        __syncthreads();                            // buf p free for reuse
    }

    // epilogue
    float sk = 1.0f, skc = 0.0f;
    if (Xold) {
        float g = skipGate ? skipGate[0] : 0.0f;
        sk  = 1.0f / (1.0f + __expf(-g));
        skc = 1.0f - sk;
    }
    #pragma unroll
    for (int sub = 0; sub < 2; ++sub) {
        #pragma unroll
        for (int j = 0; j < 4; ++j) {
            int col = n0 + j * 16 + r16;
            float bv = bias ? bias[col] : 0.0f;
            #pragma unroll
            for (int e = 0; e < 8; ++e) {
                int gm = m0 + wave * 32 + sub * 16 + hf * 8 + e;
                if (gm < M) {
                    float val = acc[sub * 4 + j][e] + bv;
                    if (reluC) val = fmaxf(val, 0.0f);
                    if (Xold)  val = sk * val + skc * Xold[(size_t)gm * N + col];
                    if (Cf) Cf[(size_t)gm * N + col] = val;
                    if (Ch) Ch[(size_t)gm * N + col] = (_Float16)val;
                }
            }
        }
    }
}

// ---------------------------------------------------------------------------
// Fold per-head relation transforms into the KQV weight; emit f16 TRANSPOSED
// combined weight Wct[768][256] (+ f32 bias bc[768]).
// Combined column layout: [0:256)=q  [256:512)=k_rel  [512:768)=v_rel.
// Reference KQV columns: [0:256)=k [256:512)=q [512:768)=v.
// ---------------------------------------------------------------------------
__global__ __launch_bounds__(256)
void prep_weights(const float* __restrict__ Wkqv, const float* __restrict__ bkqv,
                  const float* __restrict__ wk,   const float* __restrict__ wv,
                  _Float16* __restrict__ Wct, float* __restrict__ bc)
{
    int idx = blockIdx.x * 256 + threadIdx.x;
    if (idx >= 257 * 768) return;
    int row = idx / 768;                 // 256 == bias row
    int col = idx % 768;
    bool isBias = (row == 256);
    int part = col >> 8;
    int cc = col & 255;
    float r;
    if (part == 0) {
        r = isBias ? bkqv[256 + cc] : Wkqv[row * 768 + 256 + cc];
    } else {
        const float* wrel = (part == 1) ? wk : wv;
        int srcBase = (part == 1) ? 0 : 512;
        int hh = cc >> 6, ee = cc & 63;
        float acc = 0.0f;
        #pragma unroll 8
        for (int dd = 0; dd < 64; ++dd) {
            acc += (isBias ? bkqv[srcBase + hh * 64 + dd]
                           : Wkqv[row * 768 + srcBase + hh * 64 + dd])
                   * wrel[(hh * 64 + dd) * 64 + ee];
        }
        r = acc;
    }
    if (isBias) bc[col] = r;
    else        Wct[(size_t)col * 256 + row] = (_Float16)r;
}

// W[K][N] f32 -> Wt[N][K] f16
__global__ __launch_bounds__(256)
void wt_f16(const float* __restrict__ W, _Float16* __restrict__ Wt, int K, int N)
{
    int idx = blockIdx.x * 256 + threadIdx.x;
    if (idx >= K * N) return;
    int k = idx / N, c = idx % N;
    Wt[(size_t)c * K + k] = (_Float16)W[idx];
}

// f32 -> f16 elementwise (optionally exact GELU). n % 4 == 0.
__global__ __launch_bounds__(256)
void convert_f16(const float* __restrict__ in, _Float16* __restrict__ out,
                 long n, int applyGelu)
{
    long i = ((long)blockIdx.x * 256 + threadIdx.x) * 4;
    if (i >= n) return;
    float4 v = *(const float4*)(in + i);
    if (applyGelu) {
        v.x = gelu_exact(v.x); v.y = gelu_exact(v.y);
        v.z = gelu_exact(v.z); v.w = gelu_exact(v.w);
    }
    h4 o; o[0] = (_Float16)v.x; o[1] = (_Float16)v.y;
    o[2] = (_Float16)v.z; o[3] = (_Float16)v.w;
    *(h4*)(out + i) = o;
}

// ---------------------------------------------------------------------------
// Edge attention. kqv node layout (f16, stride 768):
//   [0:256)=q  [256:512)=k_rel  [512:768)=v_rel, inner index = h*64 + d.
// ---------------------------------------------------------------------------
__device__ __forceinline__ void atomicMaxF(float* addr, float val) {
    if (val >= 0.0f) atomicMax((int*)addr, __float_as_int(val));
    else             atomicMin((unsigned int*)addr, __float_as_uint(val));
}

__global__ __launch_bounds__(256)
void edge_logits(const _Float16* __restrict__ kqv_src, const _Float16* __restrict__ kqv_dst,
                 const int* __restrict__ si, const int* __restrict__ di,
                 const float* __restrict__ prior, float* __restrict__ logits,
                 float* __restrict__ mbuf, int E)
{
    int gid = blockIdx.x * 256 + threadIdx.x;
    int unit = gid >> 4;
    int lane16 = gid & 15;
    int e = unit >> 2;
    int h = unit & 3;
    if (e >= E) return;
    int s = si[e], d = di[e];
    h4 qv = *(const h4*)(kqv_dst + (size_t)d * 768 +       h * 64 + lane16 * 4);
    h4 kv = *(const h4*)(kqv_src + (size_t)s * 768 + 256 + h * 64 + lane16 * 4);
    float part = (float)qv[0] * (float)kv[0] + (float)qv[1] * (float)kv[1]
               + (float)qv[2] * (float)kv[2] + (float)qv[3] * (float)kv[3];
    part += __shfl_xor(part, 8, 16);
    part += __shfl_xor(part, 4, 16);
    part += __shfl_xor(part, 2, 16);
    part += __shfl_xor(part, 1, 16);
    if (lane16 == 0) {
        float l = part * prior[h] * 0.125f;   // 1/sqrt(64)
        logits[(size_t)e * 4 + h] = l;
        atomicMaxF(&mbuf[(size_t)d * 4 + h], l);
    }
}

__global__ __launch_bounds__(256)
void edge_expsum(const int* __restrict__ di, const float* __restrict__ mbuf,
                 float* __restrict__ logits, float* __restrict__ sbuf, int E4)
{
    int i = blockIdx.x * 256 + threadIdx.x;
    if (i >= E4) return;
    int e = i >> 2, h = i & 3;
    int d = di[e];
    float w = __expf(logits[i] - mbuf[(size_t)d * 4 + h]);
    logits[i] = w;
    atomicAdd(&sbuf[(size_t)d * 4 + h], w);
}

__global__ __launch_bounds__(256)
void edge_aggregate(const _Float16* __restrict__ kqv_src, const int* __restrict__ si,
                    const int* __restrict__ di, const float* __restrict__ wbuf,
                    const float* __restrict__ sbuf, float* __restrict__ agg, int E)
{
    int gid = blockIdx.x * 256 + threadIdx.x;
    int e = gid >> 6;
    int j = gid & 63;
    if (e >= E) return;
    int h = j >> 4;
    int d = di[e], s = si[e];
    float alpha = wbuf[(size_t)e * 4 + h] / (sbuf[(size_t)d * 4 + h] + 1e-16f);
    h4 vv = *(const h4*)(kqv_src + (size_t)s * 768 + 512 + j * 4);
    float* ap = agg + (size_t)d * 256 + j * 4;
    atomicAdd(ap + 0, alpha * (float)vv[0]);
    atomicAdd(ap + 1, alpha * (float)vv[1]);
    atomicAdd(ap + 2, alpha * (float)vv[2]);
    atomicAdd(ap + 3, alpha * (float)vv[3]);
}

__global__ __launch_bounds__(256)
void fill_u32(unsigned int* __restrict__ p, unsigned int v, int n)
{
    int i = blockIdx.x * 256 + threadIdx.x;
    if (i < n) p[i] = v;
}

// ---------------------------------------------------------------------------
// Host launcher.  Input index map (jax pytree / sorted-key flattening):
//   0 x_author, 1 x_paper,
//   2 + 16*l + {0:kqvW_a,1:kqvB_a,2:kqvW_p,3:kqvB_p,4:outW_a,5:outB_a,
//               6:outW_p,7:outB_p,8:p_ap,9:p_pa,10:skip_a,11:skip_p,
//               12:wk_ap,13:wk_pa,14:wv_ap,15:wv_pa} for l=0,1,
//   34..37 lin_in (aW,ab,pW,pb), 38 lin_out W, 39 lin_out b,
//   40 src_ap, 41 dst_ap, 42 src_pa, 43 dst_pa.
// ---------------------------------------------------------------------------
extern "C" void kernel_launch(void* const* d_in, const int* in_sizes, int n_in,
                              void* d_out, int out_size, void* d_ws, size_t ws_size,
                              hipStream_t stream)
{
    (void)in_sizes; (void)n_in; (void)out_size; (void)ws_size;
    const int NA = 100000, NP = 200000, E = 250000;

    const float* xa_in = (const float*)d_in[0];
    const float* xp_in = (const float*)d_in[1];
    const int* src_ap = (const int*)d_in[40];
    const int* dst_ap = (const int*)d_in[41];
    const int* src_pa = (const int*)d_in[42];
    const int* dst_pa = (const int*)d_in[43];

    char* ws = (char*)d_ws;
    size_t off = 0;
    auto alloc = [&](size_t bytes) -> char* {
        off = (off + 255) & ~(size_t)255;
        char* p = ws + off; off += bytes; return p;
    };
    float*    x_a    = (float*)   alloc((size_t)NA * 256 * 4);
    float*    x_p    = (float*)   alloc((size_t)NP * 256 * 4);
    _Float16* x_a_h  = (_Float16*)alloc((size_t)NA * 256 * 2);
    _Float16* x_p_h  = (_Float16*)alloc((size_t)NP * 256 * 2);
    _Float16* xa_h   = (_Float16*)alloc((size_t)NA * 256 * 2);
    _Float16* xp_h   = (_Float16*)alloc((size_t)NP * 128 * 2);
    _Float16* kqv_a  = (_Float16*)alloc((size_t)NA * 768 * 2);
    _Float16* kqv_p  = (_Float16*)alloc((size_t)NP * 768 * 2);
    float*    agg_a  = (float*)   alloc((size_t)NA * 256 * 4);
    float*    agg_p  = (float*)   alloc((size_t)NP * 256 * 4);
    _Float16* aggh_a = (_Float16*)alloc((size_t)NA * 256 * 2);
    _Float16* aggh_p = (_Float16*)alloc((size_t)NP * 256 * 2);
    float*    m_a    = (float*)   alloc((size_t)(NA + NP) * 4 * 4);
    float*    m_p    = m_a + (size_t)NA * 4;
    float*    s_a    = (float*)   alloc((size_t)(NA + NP) * 4 * 4);
    float*    s_p    = s_a + (size_t)NA * 4;
    float*    lg_ap  = (float*)   alloc((size_t)E * 4 * 4);
    float*    lg_pa  = (float*)   alloc((size_t)E * 4 * 4);
    _Float16* Wct_a  = (_Float16*)alloc(768 * 256 * 2);
    _Float16* Wct_p  = (_Float16*)alloc(768 * 256 * 2);
    float*    bc_a   = (float*)   alloc(768 * 4);
    float*    bc_p   = (float*)   alloc(768 * 4);
    _Float16* Wt_lina= (_Float16*)alloc(256 * 256 * 2);
    _Float16* Wt_linp= (_Float16*)alloc(256 * 128 * 2);
    _Float16* Wt_outa= (_Float16*)alloc(256 * 256 * 2);
    _Float16* Wt_outp= (_Float16*)alloc(256 * 256 * 2);
    _Float16* Wt_lout= (_Float16*)alloc(128 * 256 * 2);

    const int GA = (NA + BM - 1) / BM;   // 391
    const int GP = (NP + BM - 1) / BM;   // 782

    // one-time f16 conversions / weight transposes
    wt_f16<<<(256 * 256 + 255) / 256, 256, 0, stream>>>((const float*)d_in[34], Wt_lina, 256, 256);
    wt_f16<<<(128 * 256 + 255) / 256, 256, 0, stream>>>((const float*)d_in[36], Wt_linp, 128, 256);
    wt_f16<<<(256 * 128 + 255) / 256, 256, 0, stream>>>((const float*)d_in[38], Wt_lout, 256, 128);
    convert_f16<<<(int)(((long)NA * 256 / 4 + 255) / 256), 256, 0, stream>>>(xa_in, xa_h, (long)NA * 256, 0);
    convert_f16<<<(int)(((long)NP * 128 / 4 + 255) / 256), 256, 0, stream>>>(xp_in, xp_h, (long)NP * 128, 0);

    // lin_in + ReLU (writes f32 state + f16 mirror)
    gemm_wmma<<<dim3(GA, 256 / BN), 256, 0, stream>>>(
        xa_h, Wt_lina, (const float*)d_in[35], x_a, x_a_h, nullptr, nullptr, NA, 256, 256, 1);
    gemm_wmma<<<dim3(GP, 256 / BN), 256, 0, stream>>>(
        xp_h, Wt_linp, (const float*)d_in[37], x_p, x_p_h, nullptr, nullptr, NP, 256, 128, 1);

    for (int l = 0; l < 2; ++l) {
        int b = 2 + 16 * l;
        const float* kqvW_a = (const float*)d_in[b + 0];
        const float* kqvB_a = (const float*)d_in[b + 1];
        const float* kqvW_p = (const float*)d_in[b + 2];
        const float* kqvB_p = (const float*)d_in[b + 3];
        const float* outW_a = (const float*)d_in[b + 4];
        const float* outB_a = (const float*)d_in[b + 5];
        const float* outW_p = (const float*)d_in[b + 6];
        const float* outB_p = (const float*)d_in[b + 7];
        const float* p_ap   = (const float*)d_in[b + 8];
        const float* p_pa   = (const float*)d_in[b + 9];
        const float* skip_a = (const float*)d_in[b + 10];
        const float* skip_p = (const float*)d_in[b + 11];
        const float* wk_ap  = (const float*)d_in[b + 12];
        const float* wk_pa  = (const float*)d_in[b + 13];
        const float* wv_ap  = (const float*)d_in[b + 14];
        const float* wv_pa  = (const float*)d_in[b + 15];

        prep_weights<<<771, 256, 0, stream>>>(kqvW_a, kqvB_a, wk_ap, wv_ap, Wct_a, bc_a);
        prep_weights<<<771, 256, 0, stream>>>(kqvW_p, kqvB_p, wk_pa, wv_pa, Wct_p, bc_p);
        wt_f16<<<(256 * 256 + 255) / 256, 256, 0, stream>>>(outW_a, Wt_outa, 256, 256);
        wt_f16<<<(256 * 256 + 255) / 256, 256, 0, stream>>>(outW_p, Wt_outp, 256, 256);

        // fused [q | k_rel | v_rel] GEMM -> f16
        gemm_wmma<<<dim3(GA, 768 / BN), 256, 0, stream>>>(
            x_a_h, Wct_a, bc_a, nullptr, kqv_a, nullptr, nullptr, NA, 768, 256, 0);
        gemm_wmma<<<dim3(GP, 768 / BN), 256, 0, stream>>>(
            x_p_h, Wct_p, bc_p, nullptr, kqv_p, nullptr, nullptr, NP, 768, 256, 0);

        // segment-softmax state
        int nmh = (NA + NP) * 4;
        fill_u32<<<(nmh + 255) / 256, 256, 0, stream>>>((unsigned int*)m_a, 0xFF800000u, nmh);
        hipMemsetAsync(s_a, 0, (size_t)(NA + NP) * 4 * 4, stream);
        hipMemsetAsync(agg_a, 0, (size_t)NA * 256 * 4, stream);
        hipMemsetAsync(agg_p, 0, (size_t)NP * 256 * 4, stream);

        int blL = (E * 64 + 255) / 256;
        int blS = (E * 4 + 255) / 256;
        edge_logits<<<blL, 256, 0, stream>>>(kqv_a, kqv_p, src_ap, dst_ap, p_ap, lg_ap, m_p, E);
        edge_logits<<<blL, 256, 0, stream>>>(kqv_p, kqv_a, src_pa, dst_pa, p_pa, lg_pa, m_a, E);
        edge_expsum<<<blS, 256, 0, stream>>>(dst_ap, m_p, lg_ap, s_p, E * 4);
        edge_expsum<<<blS, 256, 0, stream>>>(dst_pa, m_a, lg_pa, s_a, E * 4);
        edge_aggregate<<<blL, 256, 0, stream>>>(kqv_a, src_ap, dst_ap, lg_ap, s_p, agg_p, E);
        edge_aggregate<<<blL, 256, 0, stream>>>(kqv_p, src_pa, dst_pa, lg_pa, s_a, agg_a, E);

        // gelu + f16 convert, then out-projection with sigmoid-gated skip
        convert_f16<<<(int)(((long)NA * 256 / 4 + 255) / 256), 256, 0, stream>>>(agg_a, aggh_a, (long)NA * 256, 1);
        convert_f16<<<(int)(((long)NP * 256 / 4 + 255) / 256), 256, 0, stream>>>(agg_p, aggh_p, (long)NP * 256, 1);
        gemm_wmma<<<dim3(GA, 256 / BN), 256, 0, stream>>>(
            aggh_a, Wt_outa, outB_a, x_a, x_a_h, x_a, skip_a, NA, 256, 256, 0);
        gemm_wmma<<<dim3(GP, 256 / BN), 256, 0, stream>>>(
            aggh_p, Wt_outp, outB_p, x_p, x_p_h, x_p, skip_p, NP, 256, 256, 0);
    }

    // lin_out -> d_out (author block, then paper block)
    float* out_a = (float*)d_out;
    float* out_p = out_a + (size_t)NA * 128;
    gemm_wmma<<<dim3(GA, 128 / BN), 256, 0, stream>>>(
        x_a_h, Wt_lout, (const float*)d_in[39], out_a, nullptr, nullptr, nullptr, NA, 128, 256, 0);
    gemm_wmma<<<dim3(GP, 128 / BN), 256, 0, stream>>>(
        x_p_h, Wt_lout, (const float*)d_in[39], out_p, nullptr, nullptr, nullptr, NP, 128, 256, 0);
}
